// CliffordSimplicialMPNN_2774548873295
// MI455X (gfx1250) — compile-verified, matrix operand order checked by reading, not run on hardware
//
#include <hip/hip_runtime.h>
#include <hip/hip_bf16.h>
#include <math.h>

typedef __attribute__((ext_vector_type(16))) _Float16 v16h;
typedef __attribute__((ext_vector_type(8)))  _Float16 v8h;
typedef __attribute__((ext_vector_type(8)))  float    v8f;

#define NBLD 8
#define EPSV 1e-6f
#define NPGC 5

// ---------------- Cayley table (constexpr, folds under full unroll) ----------
__host__ __device__ constexpr int blade_bit(int i){ const int t[8]={0,1,2,4,3,5,6,7}; return t[i]; }
__host__ __device__ constexpr int grade_of(int i){ const int t[8]={0,1,1,1,2,2,2,3}; return t[i]; }
__host__ __device__ constexpr int bit2idx(int b){ return b==3?4:(b==4?3:b); }
__host__ __device__ constexpr int cay_k(int i,int j){ return bit2idx(blade_bit(i)^blade_bit(j)); }
__host__ __device__ constexpr float cay_s(int i,int j){
  int a = blade_bit(i)>>1, b = blade_bit(j), s=0;
  while(a){ int m=a&b; while(m){ s+=m&1; m>>=1; } a>>=1; }
  return (s&1)? -1.f : 1.f;
}

// ---------------- WMMA helper ------------------------------------------------
__device__ inline v8f wmma16(v16h a, v16h b, v8f c){
  return __builtin_amdgcn_wmma_f32_16x16x32_f16(false, a, false, b, (short)0, c, false, false);
}

// Assemble a 16-half fragment from two contiguous 8-half LDS runs.
// ISA 7.12.2: lane l needs k' = base..base+7 and base+16..base+23, base=8*(l>=16).
__device__ inline v16h frag_from(const _Float16* rowp, int k0){
  v8h lo = *(const v8h*)(rowp + k0);
  v8h hi = *(const v8h*)(rowp + k0 + 16);
  v16h a;
  #pragma unroll
  for(int e=0;e<8;e++){ a[e] = lo[e]; a[e+8] = hi[e]; }
  return a;
}

// ---------------- Fused CEMLP block ------------------------------------------
struct CemlpArgs {
  const float* src0; const int* idx0; int c0;
  const float* src1; const int* idx1; int c1;
  int nattr, ar0, ar1, attr_first;
  const float* se;                 // [3,3] sim scalar embedding
  const float* lin_w; const float* lin_b;
  const float* silu_a; const float* silu_b;
  const float* gpl_w; const float* gpl_b;
  const float* gp_na; const float* gp_w;
  const float* ln_a;
  int Cin, Cout, nrows;
  float* out; const int* out_idx; int out_mode; // 0=store 1=atomicAdd 2=residual add
};

// KT = K tiles of GEMM1 (ceil(Cin/32)), NT = N tiles (ceil(Cout/16)).
template<int KT, int NT>
__global__ __launch_bounds__(128) void k_cemlp(CemlpArgs p){
  __shared__ __align__(16) _Float16 Ah[16][8][64];  // [row][blade][k], f16, zero-padded
  __shared__ __align__(16) _Float16 Wh[32][4][64];  // [o][grade][k],  f16, zero-padded
  __shared__ float    Xmid[16][28][8];
  __shared__ float    Pbuf[16][28][8];
  __shared__ float    nbuf[16][28];
  __shared__ float    mbuf[16];

  const int tid  = threadIdx.x;
  const int lane = tid & 31;
  const int wv   = tid >> 5;
  const int rowbase = blockIdx.x * 16;
  const int bl0 = wv*2, bl1 = wv*2+1;
  const int m_c = (lane>>4)*8;
  const int col = lane & 15;
  const int mrow = lane & 15;
  const int kbase = (lane>>4)*8;
  const int g0 = grade_of(bl0), g1 = grade_of(bl1);

  // ---- Stage 0: cooperative gather of input tile (vector global loads -> LDS f16)
  for(int idx = tid; idx < 16*64; idx += 128){
    const int m = idx >> 6, k = idx & 63;
    const int row = rowbase + m;
    float v[8] = {0,0,0,0,0,0,0,0};
    if (k < p.Cin && row < p.nrows){
      int kk = k;
      const float* bp = nullptr;
      float av = 0.f; bool attr = false;
      if (p.attr_first){
        if (kk < p.nattr){ attr=true; int a=(kk<3)?p.ar0:p.ar1; int j=(kk<3)?kk:kk-3; av=p.se[a*3+j]; }
        else {
          kk -= p.nattr;
          if (kk < p.c0){ long r = p.idx0 ? p.idx0[row] : row; bp = p.src0 + r*(long)p.c0*NBLD + kk*NBLD; }
          else { kk -= p.c0; long r = p.idx1 ? p.idx1[row] : row; bp = p.src1 + r*(long)p.c1*NBLD + kk*NBLD; }
        }
      } else {
        if (kk < p.c0){ long r = p.idx0 ? p.idx0[row] : row; bp = p.src0 + r*(long)p.c0*NBLD + kk*NBLD; }
        else if (kk < p.c0 + p.c1){ kk -= p.c0; long r = p.idx1 ? p.idx1[row] : row; bp = p.src1 + r*(long)p.c1*NBLD + kk*NBLD; }
        else { attr=true; kk -= p.c0 + p.c1; int a=(kk<3)?p.ar0:p.ar1; int j=(kk<3)?kk:kk-3; av=p.se[a*3+j]; }
      }
      if (bp){
        float4 lo = *(const float4*)bp;
        float4 hi = *(const float4*)(bp+4);
        v[0]=lo.x; v[1]=lo.y; v[2]=lo.z; v[3]=lo.w;
        v[4]=hi.x; v[5]=hi.y; v[6]=hi.z; v[7]=hi.w;
      } else if (attr){ v[0]=av; }
    }
    #pragma unroll
    for(int b=0;b<8;b++) Ah[m][b][k] = (_Float16)v[b];   // blade-major scatter
  }
  // weights for GEMM1 (4 grades contiguous in memory -> one b128 load each)
  for(int idx = tid; idx < 32*64; idx += 128){
    const int o = idx >> 6, k = idx & 63;
    float4 w = make_float4(0.f,0.f,0.f,0.f);
    if (o < p.Cout && k < p.Cin) w = *(const float4*)(p.lin_w + ((long)o*p.Cin + k)*4);
    Wh[o][0][k]=(_Float16)w.x; Wh[o][1][k]=(_Float16)w.y;
    Wh[o][2][k]=(_Float16)w.z; Wh[o][3][k]=(_Float16)w.w;
  }
  __syncthreads();

  // ---- Phase A: WMMA GEMM1 (2x ds_load_b128 per fragment, no divergence) ----
  v8f acc[2][NT];
  #pragma unroll
  for(int bi=0;bi<2;bi++)
    #pragma unroll
    for(int nt=0;nt<NT;nt++)
      #pragma unroll
      for(int q=0;q<8;q++) acc[bi][nt][q]=0.f;

  #pragma unroll
  for(int kt=0; kt<KT; kt++){
    const int k0 = kt*32 + kbase;
    v16h A0 = frag_from(&Ah[mrow][bl0][0], k0);
    v16h A1 = frag_from(&Ah[mrow][bl1][0], k0);
    #pragma unroll
    for(int nt=0; nt<NT; nt++){
      const int o = nt*16 + col;
      v16h B0 = frag_from(&Wh[o][g0][0], k0);
      v16h B1 = frag_from(&Wh[o][g1][0], k0);  // unconditional: same-addr reload is cheaper than exec dance
      acc[0][nt] = wmma16(A0, B0, acc[0][nt]);
      acc[1][nt] = wmma16(A1, B1, acc[1][nt]);
    }
  }
  // bias on scalar blade (hoisted load), spill C frags to LDS
  #pragma unroll
  for(int nt=0; nt<NT; nt++){
    const int o = nt*16 + col;
    const float bias = (o < p.Cout) ? p.lin_b[o] : 0.f;   // one guarded load per nt
    if (o < p.Cout){
      #pragma unroll
      for(int bi=0;bi<2;bi++){
        const int blade = bl0+bi;
        const float badd = (blade==0) ? bias : 0.f;       // select, no load in loop
        #pragma unroll
        for(int r=0;r<8;r++){
          Xmid[r + m_c][o][blade] = acc[bi][nt][r] + badd;
        }
      }
    }
  }
  __syncthreads();

  // ---- Phase B: restage gpl_w; mvsilu + geometric product; x2 -> Ah (f16) ----
  for(int idx = tid; idx < 32*64; idx += 128){
    const int o = idx >> 6, k = idx & 63;
    float4 w = make_float4(0.f,0.f,0.f,0.f);
    if (o < p.Cout && k < p.Cout) w = *(const float4*)(p.gpl_w + ((long)o*p.Cout + k)*4);
    Wh[o][0][k]=(_Float16)w.x; Wh[o][1][k]=(_Float16)w.y;
    Wh[o][2][k]=(_Float16)w.z; Wh[o][3][k]=(_Float16)w.w;
  }
  {  // zero-pad Ah k in [Cout,32) for GEMM2's A operand
    const int span = 32 - p.Cout;
    for(int idx = tid; idx < 16*8*span; idx += 128){
      const int m = idx / (8*span);
      const int rem = idx % (8*span);
      const int b = rem / span;
      const int k = p.Cout + rem % span;
      Ah[m][b][k] = (_Float16)0.f;
    }
  }
  for(int idx = tid; idx < 16*p.Cout; idx += 128){
    const int m = idx / p.Cout, c = idx % p.Cout;
    float y[8];
    #pragma unroll
    for(int b=0;b<8;b++) y[b] = Xmid[m][c][b];
    float q[4] = {0,0,0,0};
    #pragma unroll
    for(int b=0;b<8;b++) q[grade_of(b)] += y[b]*y[b];
    float ns[4];
    #pragma unroll
    for(int g=0;g<4;g++) ns[g] = sqrtf(q[g] + EPSV);
    ns[0] = y[0];   // scalar blade replaces grade-0 norm
    float gate[4];
    #pragma unroll
    for(int g=0;g<4;g++){
      float z = p.silu_a[c*4+g]*ns[g] + p.silu_b[c*4+g];
      gate[g] = 1.f/(1.f+__expf(-z));
    }
    float x2[8];
    #pragma unroll
    for(int b=0;b<8;b++) x2[b] = y[b]*gate[grade_of(b)];
    float q2[4]={0,0,0,0};
    #pragma unroll
    for(int b=0;b<8;b++) q2[grade_of(b)] += x2[b]*x2[b];
    float rn[4];
    #pragma unroll
    for(int g=0;g<4;g++){
      float s = 1.f/(1.f+__expf(-p.gp_na[c*4+g]));
      rn[g] = 1.f/(s*(sqrtf(q2[g]+EPSV)-1.f)+1.f);
    }
    float xr[8];
    #pragma unroll
    for(int b=0;b<8;b++) xr[b] = x2[b]*rn[grade_of(b)];
    float prod[8]={0,0,0,0,0,0,0,0};
    #pragma unroll
    for(int i=0;i<8;i++){
      #pragma unroll
      for(int j=0;j<8;j++){
        const int k = cay_k(i,j);
        const float sg = cay_s(i,j);
        prod[k] += sg * p.gp_w[c*64 + grade_of(i)*16 + grade_of(j)*4 + grade_of(k)] * x2[i]*xr[j];
      }
    }
    #pragma unroll
    for(int b=0;b<8;b++){ Ah[m][b][c] = (_Float16)x2[b]; Pbuf[m][c][b] = prod[b]; }
  }
  __syncthreads();

  // ---- Phase C: WMMA GEMM2 (gp.left linear, K = Cout <= 28, one K tile) ----
  v8f acc2[2][NT];
  #pragma unroll
  for(int bi=0;bi<2;bi++)
    #pragma unroll
    for(int nt=0;nt<NT;nt++)
      #pragma unroll
      for(int q=0;q<8;q++) acc2[bi][nt][q]=0.f;
  {
    v16h A0 = frag_from(&Ah[mrow][bl0][0], kbase);
    v16h A1 = frag_from(&Ah[mrow][bl1][0], kbase);
    #pragma unroll
    for(int nt=0; nt<NT; nt++){
      const int o = nt*16 + col;
      v16h B0 = frag_from(&Wh[o][g0][0], kbase);
      v16h B1 = frag_from(&Wh[o][g1][0], kbase);
      acc2[0][nt] = wmma16(A0, B0, acc2[0][nt]);
      acc2[1][nt] = wmma16(A1, B1, acc2[1][nt]);
    }
  }
  const float is2 = 0.70710678f;
  #pragma unroll
  for(int nt=0; nt<NT; nt++){
    const int o = nt*16 + col;
    const float bias = (o < p.Cout) ? p.gpl_b[o] : 0.f;   // hoisted guarded load
    if (o < p.Cout){
      #pragma unroll
      for(int bi=0;bi<2;bi++){
        const int blade = bl0+bi;
        const float badd = (blade==0) ? bias : 0.f;
        #pragma unroll
        for(int r=0;r<8;r++){
          const int m = r + m_c;
          Xmid[m][o][blade] = (acc2[bi][nt][r] + badd + Pbuf[m][o][blade]) * is2;
        }
      }
    }
  }
  __syncthreads();

  // ---- Phase D: mvlayernorm + output (store / residual / atomic scatter) ----
  for(int idx = tid; idx < 16*p.Cout; idx += 128){
    const int m = idx / p.Cout, c = idx % p.Cout;
    float s=0;
    #pragma unroll
    for(int b=0;b<8;b++){ float v=Xmid[m][c][b]; s+=v*v; }
    nbuf[m][c] = sqrtf(s + EPSV);
  }
  __syncthreads();
  if (tid < 16){
    float s=0;
    for(int c=0;c<p.Cout;c++) s += nbuf[tid][c];
    mbuf[tid] = s / (float)p.Cout;
  }
  __syncthreads();
  for(int idx = tid; idx < 16*p.Cout*8; idx += 128){
    const int b = idx & 7;
    const int t = idx >> 3;
    const int c = t % p.Cout;
    const int m = t / p.Cout;
    const float v = p.ln_a[c] * Xmid[m][c][b] / (mbuf[m] + EPSV);
    const long row = rowbase + m;
    if (row < p.nrows){
      const long orow = p.out_idx ? (long)p.out_idx[row] : row;
      const long off = orow*(long)p.Cout*8 + c*8 + b;
      if      (p.out_mode==1) atomicAdd(&p.out[off], v);
      else if (p.out_mode==2) p.out[off] += v;
      else                    p.out[off]  = v;
    }
  }
}

// ---------------- small helper kernels ---------------------------------------
__global__ void k_zero(float* p, long n){
  long i = (long)blockIdx.x*blockDim.x + threadIdx.x;
  long st = (long)gridDim.x*blockDim.x;
  for(; i<n; i+=st) p[i]=0.f;
}

__global__ void k_se(const float* emb, const float* W, const float* b, float* se){
  int t = threadIdx.x;
  if (t < 9){
    int i=t/3, j=t%3;
    float s=b[j];
    for(int h=0;h<28;h++) s += emb[i*28+h]*W[j*28+h];
    se[t]=s;
  }
}

__global__ void k_featin(const float* loc, const float* vel, const float* chg, float* fi, int N){
  int n = blockIdx.x*blockDim.x + threadIdx.x;
  if (n>=N) return;
  int g = n/NPGC;
  float mx=0,my=0,mz=0;
  for(int v=0;v<NPGC;v++){ const float* l=&loc[(g*NPGC+v)*3]; mx+=l[0];my+=l[1];mz+=l[2]; }
  mx*=0.2f; my*=0.2f; mz*=0.2f;
  float* o = &fi[(long)n*24];
  for(int k=0;k<24;k++) o[k]=0.f;
  o[0] = chg[n];                                // charge -> scalar blade, ch0
  o[8+1]=loc[n*3+0]-mx; o[8+2]=loc[n*3+1]-my; o[8+3]=loc[n*3+2]-mz;  // ch1 e1..e3
  o[16+1]=vel[n*3+0];   o[16+2]=vel[n*3+1];   o[16+3]=vel[n*3+2];    // ch2 e1..e3
}

__global__ void k_vavg(const float* fi, const int* xi, const int* bat, int nv, int nrows, float* out){
  int r = blockIdx.x*blockDim.x + threadIdx.x;
  if (r>=nrows) return;
  float inv = 1.f/(float)nv;
  for(int c=0;c<24;c++){
    float s=0;
    for(int v=0;v<nv;v++){
      long node = (long)xi[r*nv+v] + (long)bat[r]*NPGC;
      s += fi[node*24 + c];
    }
    out[(long)r*24+c] = s*inv;
  }
}

__global__ void k_predloss(const float* loc, const float* y, const float* o1,
                           float* out, float* acc, int N){
  int n = blockIdx.x*blockDim.x + threadIdx.x;
  if (n>=N) return;
  float s=0;
  for(int j=0;j<3;j++){
    float lp = loc[n*3+j] + o1[(long)n*8 + 1 + j];
    out[1 + n*3 + j] = lp;
    float d = lp - y[n*3+j];
    s += d*d;
  }
  atomicAdd(acc, s);
}

__global__ void k_finloss(float* out, const float* acc, float inv){
  if (threadIdx.x==0) out[0] = acc[0]*inv;
}

// ---------------- host side --------------------------------------------------
// jax sorted-key tree-flatten order of setup_inputs():
//  adj(6) charges eaa(6) loc params(...) vel x_0 x_0b x_1 x_1b x_2 x_2b y
#define IN_ADJ(t)   (t)         // '0_0','0_1','1_0','1_1','1_2','2_1'
#define IN_CHARGES  6
#define IN_LOC      13
#define PRM_FEAT    14          // 1 cemlp block = 9 leaves
#define PRM_LAYER(L) (23 + 81*(L))
#define PRM_MSG(L,t) (PRM_LAYER(L) + 9*(t))
#define PRM_UPD(L,d) (PRM_LAYER(L) + 54 + 9*(d))
#define PRM_OUT0    266
#define PRM_OUT1    275
#define PRM_SIMW    284
#define PRM_SIMB    285
#define PRM_SIMEMB  286
#define IN_VEL      287
#define IN_X0       288
#define IN_X0B      289
#define IN_X1       290
#define IN_X1B      291
#define IN_X2       292
#define IN_X2B      293
#define IN_Y        294

struct BP { const float *gpl_b,*gpl_w,*gp_na,*gp_w,*lin_b,*lin_w,*ln_a,*silu_a,*silu_b; };
// block leaf order (sorted keys): gp.left.b gp.left.w gp.norm_a gp.w lin.b lin.w ln.a silu.a silu.b
static BP blockAt(void* const* din, int base){
  BP b;
  b.gpl_b =(const float*)din[base+0]; b.gpl_w =(const float*)din[base+1];
  b.gp_na =(const float*)din[base+2]; b.gp_w  =(const float*)din[base+3];
  b.lin_b =(const float*)din[base+4]; b.lin_w =(const float*)din[base+5];
  b.ln_a  =(const float*)din[base+6]; b.silu_a=(const float*)din[base+7];
  b.silu_b=(const float*)din[base+8];
  return b;
}
static void fillBP(CemlpArgs& a, const BP& b){
  a.gpl_b=b.gpl_b; a.gpl_w=b.gpl_w; a.gp_na=b.gp_na; a.gp_w=b.gp_w;
  a.lin_b=b.lin_b; a.lin_w=b.lin_w; a.ln_a=b.ln_a; a.silu_a=b.silu_a; a.silu_b=b.silu_b;
}

// static-shape dispatch: keeps WMMA accumulators statically register-allocated
static void launch_cemlp(const CemlpArgs& a, hipStream_t stream){
  const int blocks = (a.nrows + 15)/16;
  const int kt = (a.Cin + 31)/32, nt = (a.Cout + 15)/16;
  if (kt==2 && nt==2)      k_cemlp<2,2><<<blocks,128,0,stream>>>(a);
  else if (kt==1 && nt==2) k_cemlp<1,2><<<blocks,128,0,stream>>>(a);
  else                     k_cemlp<1,1><<<blocks,128,0,stream>>>(a);
}

extern "C" void kernel_launch(void* const* d_in, const int* in_sizes, int n_in,
                              void* d_out, int out_size, void* d_ws, size_t ws_size,
                              hipStream_t stream){
  (void)n_in; (void)out_size; (void)ws_size;
  const float* loc = (const float*)d_in[IN_LOC];
  const float* vel = (const float*)d_in[IN_VEL];
  const float* chg = (const float*)d_in[IN_CHARGES];
  const float* yv  = (const float*)d_in[IN_Y];
  const int N  = in_sizes[IN_CHARGES];       // 5120 nodes
  const int n0 = in_sizes[IN_X0];            // [n0,1]
  const int n1 = in_sizes[IN_X1]/2;          // [n1,2]
  const int n2 = in_sizes[IN_X2]/3;          // [n2,3]
  const int nr[3] = {n0,n1,n2};

  float* ws = (float*)d_ws;
  long off=0;
  float* se  = ws+off; off+=16;
  float* acc = ws+off; off+=16;
  float* fi  = ws+off; off+=(long)N*24;
  float* vin[3];
  vin[0]=ws+off; off+=(long)n0*24;
  vin[1]=ws+off; off+=(long)n1*24;
  vin[2]=ws+off; off+=(long)n2*24;
  float* xs[3];
  xs[0]=ws+off; off+=(long)n0*224;
  xs[1]=ws+off; off+=(long)n1*224;
  xs[2]=ws+off; off+=(long)n2*224;
  float* aggbase = ws+off;
  float* ag[3];
  ag[0]=ws+off; off+=(long)n0*224;
  ag[1]=ws+off; off+=(long)n1*224;
  ag[2]=ws+off; off+=(long)n2*224;
  const long aggcnt = (long)(n0+n1+n2)*224;
  float* o1 = ws+off; off+=(long)n0*8;

  // sim-type embedding se[3][3], feat_in, per-simplex vertex averages
  k_se<<<1,16,0,stream>>>((const float*)d_in[PRM_SIMEMB], (const float*)d_in[PRM_SIMW],
                          (const float*)d_in[PRM_SIMB], se);
  k_featin<<<(N+127)/128,128,0,stream>>>(loc, vel, chg, fi, N);
  k_vavg<<<(n0+127)/128,128,0,stream>>>(fi,(const int*)d_in[IN_X0],(const int*)d_in[IN_X0B],1,n0,vin[0]);
  k_vavg<<<(n1+127)/128,128,0,stream>>>(fi,(const int*)d_in[IN_X1],(const int*)d_in[IN_X1B],2,n1,vin[1]);
  k_vavg<<<(n2+127)/128,128,0,stream>>>(fi,(const int*)d_in[IN_X2],(const int*)d_in[IN_X2B],3,n2,vin[2]);

  // feat_emb cemlp: [sim(3) | avg-feat(3)] -> 28
  BP fb = blockAt(d_in, PRM_FEAT);
  for(int d=0;d<3;d++){
    CemlpArgs a={};
    a.src0=vin[d]; a.idx0=nullptr; a.c0=3; a.src1=nullptr; a.idx1=nullptr; a.c1=0;
    a.nattr=3; a.ar0=d; a.ar1=d; a.attr_first=1; a.se=se;
    fillBP(a,fb);
    a.Cin=6; a.Cout=28; a.nrows=nr[d];
    a.out=xs[d]; a.out_idx=nullptr; a.out_mode=0;
    launch_cemlp(a, stream);
  }

  // message-passing layers
  const int SI[6]={0,0,1,1,1,2}, SJ[6]={0,1,0,1,2,1};  // sorted adj key order
  for(int L=0;L<3;L++){
    k_zero<<<1024,256,0,stream>>>(aggbase, aggcnt);
    for(int t=0;t<6;t++){
      const int E = in_sizes[IN_ADJ(t)]/2;
      const int* adj = (const int*)d_in[IN_ADJ(t)];
      CemlpArgs a={};
      a.src0=xs[SI[t]]; a.idx0=adj;     a.c0=28;   // x_si[send]
      a.src1=xs[SJ[t]]; a.idx1=adj+E;   a.c1=28;   // x_sj[recv]
      a.nattr=6; a.ar0=SI[t]; a.ar1=SJ[t]; a.attr_first=0; a.se=se;
      fillBP(a, blockAt(d_in, PRM_MSG(L,t)));
      a.Cin=62; a.Cout=28; a.nrows=E;
      a.out=ag[SJ[t]]; a.out_idx=adj+E; a.out_mode=1;   // segment_sum via atomics
      launch_cemlp(a, stream);
    }
    for(int d=0;d<3;d++){
      CemlpArgs a={};
      a.src0=xs[d]; a.idx0=nullptr; a.c0=28;
      a.src1=ag[d]; a.idx1=nullptr; a.c1=28;
      a.nattr=3; a.ar0=d; a.ar1=d; a.attr_first=0; a.se=se;
      fillBP(a, blockAt(d_in, PRM_UPD(L,d)));
      a.Cin=59; a.Cout=28; a.nrows=nr[d];
      a.out=xs[d]; a.out_idx=nullptr; a.out_mode=2;     // residual x += upd(...)
      launch_cemlp(a, stream);
    }
  }

  // output MLP on 0-simplices: 28 -> 28 -> 1 (reuse ag[0] as temp)
  {
    CemlpArgs a={};
    a.src0=xs[0]; a.idx0=nullptr; a.c0=28; a.src1=nullptr; a.idx1=nullptr; a.c1=0;
    a.nattr=0; a.ar0=0; a.ar1=0; a.attr_first=0; a.se=se;
    fillBP(a, blockAt(d_in, PRM_OUT0));
    a.Cin=28; a.Cout=28; a.nrows=n0;
    a.out=ag[0]; a.out_idx=nullptr; a.out_mode=0;
    launch_cemlp(a, stream);
  }
  {
    CemlpArgs a={};
    a.src0=ag[0]; a.idx0=nullptr; a.c0=28; a.src1=nullptr; a.idx1=nullptr; a.c1=0;
    a.nattr=0; a.ar0=0; a.ar1=0; a.attr_first=0; a.se=se;
    fillBP(a, blockAt(d_in, PRM_OUT1));
    a.Cin=28; a.Cout=1; a.nrows=n0;
    a.out=o1; a.out_idx=nullptr; a.out_mode=0;
    launch_cemlp(a, stream);
  }

  // loss + loc_pred -> d_out = [loss, loc_pred(N*3)]
  k_zero<<<1,32,0,stream>>>(acc, 1);
  k_predloss<<<(N+127)/128,128,0,stream>>>(loc, yv, o1, (float*)d_out, acc, N);
  k_finloss<<<1,1,0,stream>>>((float*)d_out, acc, 1.0f/(float)(N*3));
}
